// FreeConvNetwork_704374636650
// MI455X (gfx1250) — matrix operand order repaired and checked by reading, not compiled, not run on hardware
//
#include <hip/hip_runtime.h>
#include <hip/hip_bf16.h>

typedef __bf16 bf16;
typedef __attribute__((ext_vector_type(16))) __bf16 v16bf;
typedef __attribute__((ext_vector_type(8)))  __bf16 v8bf;
typedef __attribute__((ext_vector_type(8)))  float  v8f;

// ---------------------------------------------------------------------------
// Layer 1: 1->32, 28x28 -> 13x13, stride 2, K=9. Memory-bound; fp32 VALU,
// store bf16 activations for the WMMA layers.
// One thread per output element, thread id == h1 flat index (coalesced store).
// ---------------------------------------------------------------------------
__global__ __launch_bounds__(256) void lc1_kernel(const float* __restrict__ x,
                                                  const float* __restrict__ W1,
                                                  const float* __restrict__ b1,
                                                  bf16* __restrict__ h1) {
    int t  = blockIdx.x * 256 + threadIdx.x;   // t = (b*32 + o)*169 + ij
    int ij = t % 169;
    int bo = t / 169;
    int o  = bo & 31;
    int b  = bo >> 5;
    int i = ij / 13, j = ij % 13;
    const float* xp = x + b * 784 + (2 * i) * 28 + (2 * j);
    float acc = b1[o * 169 + ij];
#pragma unroll
    for (int dy = 0; dy < 3; ++dy)
#pragma unroll
        for (int dx = 0; dx < 3; ++dx)
            acc = fmaf(xp[dy * 28 + dx], W1[(o * 9 + dy * 3 + dx) * 169 + ij], acc);
    h1[t] = (bf16)fmaxf(acc, 0.0f);
}

// ---------------------------------------------------------------------------
// Generic locally-connected layer as per-position GEMM on the bf16 WMMA pipe.
// Block = one output spatial position s and MB batch rows.
//   LDS Wl[o][k]  : weight slice for s (k-contiguous rows -> b128 frag loads)
//   LDS Al[m][k]  : im2col patches      (k-contiguous rows -> b128 frag loads)
// Each wave owns a 16-row M tile, iterates all O/16 N tiles, K in steps of 32.
// Fragment layouts follow CDNA5 ISA §7.12.2 (wave32, 16-bit A/B).
// ---------------------------------------------------------------------------
template <int CIN, int HIN, int WIN, int O, int OH, int OW, int STRIDE,
          int MB, int NTHREADS>
__global__ __launch_bounds__(NTHREADS) void lc_wmma_kernel(
    const bf16*  __restrict__ in,   // (B, CIN, HIN, WIN) bf16
    const float* __restrict__ Wg,   // (O, CIN*9, OH, OW) fp32
    const float* __restrict__ bg,   // (O, OH, OW) fp32
    bf16*        __restrict__ out)  // (B, O, OH, OW) bf16, post-ReLU
{
    constexpr int K  = CIN * 9;
    constexpr int S  = OH * OW;
    constexpr int NT = O / 16;

    __shared__ alignas(16) bf16 Wl[O * K];
    __shared__ alignas(16) bf16 Al[MB * K];

    const int s    = blockIdx.x % S;
    const int mblk = blockIdx.x / S;
    const int i    = s / OW, j = s % OW;
    const int tid  = threadIdx.x;
    const int bbase = mblk * MB;

    // Stage weight slice for this position: Wl[o*K + k] = Wg[(o*K+k)*S + s]
    for (int e = tid; e < O * K; e += NTHREADS)
        Wl[e] = (bf16)Wg[e * S + s];

    // Stage im2col patches: Al[m*K + (c*9 + dy*3 + dx)]
    for (int e = tid; e < MB * K; e += NTHREADS) {
        int m = e / K, k = e % K;
        int c = k / 9, r = k % 9;
        int dy = r / 3, dx = r % 3;
        Al[e] = in[((bbase + m) * CIN + c) * (HIN * WIN)
                   + (STRIDE * i + dy) * WIN + (STRIDE * j + dx)];
    }
    __syncthreads();

    const int wave  = tid >> 5;
    const int lane  = tid & 31;
    const int laneN = lane & 15;
    const int hiA   = (lane < 16) ? 0 : 8;    // A: K offset per half-wave
    const int hiB   = (lane < 16) ? 0 : 8;    // B: K offset per half-wave
    const int m0    = wave * 16;

    v8f acc[NT] = {};

    const bf16* arow = &Al[(m0 + laneN) * K];
    for (int k0 = 0; k0 < K; k0 += 32) {
        union { v16bf v; v8bf h[2]; } au;
        au.h[0] = *(const v8bf*)(arow + k0 + hiA);        // K = hiA..hiA+7
        au.h[1] = *(const v8bf*)(arow + k0 + hiA + 16);   // K = hiA+16..hiA+23
#pragma unroll
        for (int nt = 0; nt < NT; ++nt) {
            const bf16* brow = &Wl[(nt * 16 + laneN) * K + k0 + hiB];
            union { v16bf v; v8bf h[2]; } bu;
            bu.h[0] = *(const v8bf*)(brow);               // K = hiB..hiB+7
            bu.h[1] = *(const v8bf*)(brow + 16);          // K = hiB+16..hiB+23
            acc[nt] = __builtin_amdgcn_wmma_f32_16x16x32_bf16(
                false, au.v, false, bu.v, (short)0, acc[nt], false, false);
        }
    }

    // C layout: lanes 0-15 -> M = v, lanes 16-31 -> M = v+8; N = lane&15.
#pragma unroll
    for (int nt = 0; nt < NT; ++nt) {
        int o = nt * 16 + laneN;
        float bias = bg[o * S + s];
#pragma unroll
        for (int v = 0; v < 8; ++v) {
            int m = m0 + ((lane < 16) ? v : v + 8);
            float val = acc[nt][v] + bias;
            out[((bbase + m) * O + o) * S + s] = (bf16)fmaxf(val, 0.0f);
        }
    }
}

// ---------------------------------------------------------------------------
// Final linear layer: (8192,2048) @ (2048,10)^T + b, ReLU, fp32 out.
// N padded to a single 16-wide WMMA tile (rows 10..15 of W zeroed in LDS).
// A rows are K-contiguous in global (channel-major flatten == h3 layout),
// so A fragments load straight from global memory.
// ---------------------------------------------------------------------------
__global__ __launch_bounds__(256) void linear_kernel(
    const bf16*  __restrict__ h3,   // (B, 2048) bf16
    const float* __restrict__ Wlg,  // (10, 2048) fp32
    const float* __restrict__ blg,  // (10,) fp32
    float*       __restrict__ outp) // (B, 10) fp32
{
    constexpr int KD = 2048;
    __shared__ alignas(16) bf16 Wl[16 * KD];

    const int tid = threadIdx.x;
    for (int e = tid; e < 16 * KD; e += 256) {
        int n = e >> 11;                       // e = n*2048 + k
        Wl[e] = (n < 10) ? (bf16)Wlg[e] : (bf16)0.0f;
    }
    __syncthreads();

    const int wave  = tid >> 5;
    const int lane  = tid & 31;
    const int laneN = lane & 15;
    const int hiA   = (lane < 16) ? 0 : 8;
    const int hiB   = (lane < 16) ? 0 : 8;
    const int mbase = blockIdx.x * 128 + wave * 16;

    v8f acc = {};
    const bf16* arow = h3 + (size_t)(mbase + laneN) * KD;
    const bf16* brow = &Wl[laneN * KD];
    for (int k0 = 0; k0 < KD; k0 += 32) {
        union { v16bf v; v8bf h[2]; } au, bu;
        au.h[0] = *(const v8bf*)(arow + k0 + hiA);
        au.h[1] = *(const v8bf*)(arow + k0 + hiA + 16);
        bu.h[0] = *(const v8bf*)(brow + k0 + hiB);
        bu.h[1] = *(const v8bf*)(brow + k0 + hiB + 16);
        acc = __builtin_amdgcn_wmma_f32_16x16x32_bf16(
            false, au.v, false, bu.v, (short)0, acc, false, false);
    }

    if (laneN < 10) {                          // divergence only after all WMMAs
        float bias = blg[laneN];
#pragma unroll
        for (int v = 0; v < 8; ++v) {
            int m = mbase + ((lane < 16) ? v : v + 8);
            outp[m * 10 + laneN] = fmaxf(acc[v] + bias, 0.0f);
        }
    }
}

// ---------------------------------------------------------------------------
// Launch: 4 kernels, bf16 intermediates in workspace.
//   h1: 8192*32*169 bf16 =  88,604,672 B
//   h2: 8192*64*36  bf16 =  37,748,736 B
//   h3: 8192*2048   bf16 =  33,554,432 B   (total ~160 MB)
// ---------------------------------------------------------------------------
extern "C" void kernel_launch(void* const* d_in, const int* in_sizes, int n_in,
                              void* d_out, int out_size, void* d_ws, size_t ws_size,
                              hipStream_t stream) {
    (void)in_sizes; (void)n_in; (void)out_size; (void)ws_size;
    const float* x  = (const float*)d_in[0];
    const float* W1 = (const float*)d_in[1];
    const float* b1 = (const float*)d_in[2];
    const float* W2 = (const float*)d_in[3];
    const float* b2 = (const float*)d_in[4];
    const float* W3 = (const float*)d_in[5];
    const float* b3 = (const float*)d_in[6];
    const float* Wf = (const float*)d_in[7];
    const float* bf = (const float*)d_in[8];
    float* out = (float*)d_out;

    char* ws = (char*)d_ws;
    bf16* h1 = (bf16*)(ws);
    bf16* h2 = (bf16*)(ws + 88604672ull);
    bf16* h3 = (bf16*)(ws + 88604672ull + 37748736ull);

    // LC1: 8192*32*169 outputs / 256 = 173056 blocks
    lc1_kernel<<<173056, 256, 0, stream>>>(x, W1, b1, h1);

    // LC2: 36 positions x (8192/128) m-blocks; GEMM 8192x64x288 per position
    lc_wmma_kernel<32, 13, 13, 64, 6, 6, 2, 128, 256>
        <<<36 * 64, 256, 0, stream>>>(h1, W2, b2, h2);

    // LC3: 16 positions x (8192/64) m-blocks; GEMM 8192x128x576 per position
    lc_wmma_kernel<64, 6, 6, 128, 4, 4, 1, 64, 128>
        <<<16 * 128, 128, 0, stream>>>(h2, W3, b3, h3);

    // Linear: 8192 rows / 128 per block
    linear_kernel<<<64, 256, 0, stream>>>(h3, Wf, bf, out);
}
